// LiquidNeuralNetwork_14491219656708
// MI455X (gfx1250) — matrix operand order repaired and checked by reading, not compiled, not run on hardware
//
#include <hip/hip_runtime.h>
#include <cstddef>

typedef __attribute__((ext_vector_type(16))) _Float16 v16h;
typedef __attribute__((ext_vector_type(8)))  _Float16 v8h;
typedef __attribute__((ext_vector_type(8)))  float    v8f;

#define T_STEPS 1024
#define IN_F    128
#define U0      135
#define U1      89
#define U2      32
#define OUTF    32

// K (cat) padded to multiples of 32 for the WMMA K-loop
#define K0 288   // 263 -> 288
#define K1 224   // exact
#define K2 128   // 121 -> 128
#define KF 32    // exact

#define NT1 6
#define NT2 2
#define NTF 2
#define NKC0 9
#define NKC1 7
#define NKC2 4

#define FRAG_HALVES 512   // one 16x32 f16 B-fragment: 32 lanes * 16 halves
#define NTHREADS 288      // 9 waves (wave32)

// ---------------------------------------------------------------------------
// Prep: f32 weights (optionally NCP-masked) -> f16 WMMA-B fragment layout.
// B operand (W is [units, cat], D[m,n] = sum_k z[m,k]*W[n,k]):
//   lane l: column n = tile*16 + (l&15); halves e=0..15 map to
//   K = kc*32 + (l>>4)*8 + (e&7) + (e>>3)*16  (matches A-fragment K order).
// Zero-padded past units/cat so ghost tiles contribute exactly zero.
// ---------------------------------------------------------------------------
__global__ void lnn_prep(const float* __restrict__ W, const int* __restrict__ mask,
                         _Float16* __restrict__ dst, int units, int cat,
                         int ntiles, int nkc)
{
  int idx = blockIdx.x * blockDim.x + threadIdx.x;
  int total = ntiles * nkc * FRAG_HALVES;
  if (idx >= total) return;
  int e    = idx & 15;
  int lane = (idx >> 4) & 31;
  int f    = idx >> 9;
  int kc   = f % nkc;
  int tl   = f / nkc;
  int n    = tl * 16 + (lane & 15);
  int g    = lane >> 4;
  int k    = kc * 32 + g * 8 + (e & 7) + ((e >> 3) << 4);
  float v = 0.0f;
  if (n < units && k < cat) {
    v = W[(size_t)n * cat + k];
    if (mask) v *= (float)mask[(size_t)n * cat + k];
  }
  dst[idx] = (_Float16)v;
}

struct LNNParams {
  const float*    x;
  const _Float16* w[13];   // l0 ff1,ff2,ta,tb | l1 x4 | l2 x4 | fc
  const float*    b[13];
  float*          out;
};

__device__ __forceinline__ v8f wmma16(v16h a, v16h b, v8f c) {
  // D = A(16x32 f16) * B(32x16 f16) + C(16x16 f32)
  return __builtin_amdgcn_wmma_f32_16x16x32_f16(false, a, false, b, (short)0, c,
                                                false, false);
}

// A-fragment from a row-major f16 LDS buffer, per ISA 16-bit A 16x32 layout:
// lane group g=l>>4: halves e<8 -> K = kc*32 + e + 8g ; e>=8 -> K = kc*32 + 8 + e + 8g
__device__ __forceinline__ v16h load_a_frag(const _Float16* zrow, int kc, int g) {
  const _Float16* zp = zrow + kc * 32 + g * 8;
  v8h lo = *(const v8h*)zp;          // K = kc*32 + 8g .. +7
  v8h hi = *(const v8h*)(zp + 16);   // K = kc*32 + 16 + 8g .. +7
  return __builtin_shufflevector(lo, hi, 0,1,2,3,4,5,6,7,8,9,10,11,12,13,14,15);
}

template<int NKC, int ZSTR>
__device__ __forceinline__ void tile_mm4(const _Float16* z,
                                         const _Float16* w0, const _Float16* w1,
                                         const _Float16* w2, const _Float16* w3,
                                         int lane, v8f acc[4])
{
  const int r = lane & 15;
  const int g = lane >> 4;
  const _Float16* zrow = z + r * ZSTR;
#pragma unroll
  for (int kc = 0; kc < NKC; ++kc) {
    v16h a = load_a_frag(zrow, kc, g);
    const int wo = (kc * 32 + lane) * 16;   // per-lane 32B coalesced fragment
    acc[0] = wmma16(a, *(const v16h*)(w0 + wo), acc[0]);
    acc[1] = wmma16(a, *(const v16h*)(w1 + wo), acc[1]);
    acc[2] = wmma16(a, *(const v16h*)(w2 + wo), acc[2]);
    acc[3] = wmma16(a, *(const v16h*)(w3 + wo), acc[3]);
  }
}

__device__ __forceinline__ float sigmoidf_(float x) {
  return 1.0f / (1.0f + __expf(-x));
}

// ---------------------------------------------------------------------------
// Persistent recurrent kernel: 16 WGs x 16 batch rows, 9 waves each.
// Double-buffered LDS activation rows (parity p reads, writes go where the
// next consumer reads) remove all within-phase RAW/WAR hazards; 3 barriers
// per timestep. FC (phase D) overlaps the next step's layer0.
// ---------------------------------------------------------------------------
__global__ __launch_bounds__(NTHREADS, 1) void lnn_main(LNNParams P)
{
  __shared__ __align__(16) _Float16 z0[2][16 * K0];  // [x | h0]
  __shared__ __align__(16) _Float16 z1[2][16 * K1];  // [h0 | h1]
  __shared__ __align__(16) _Float16 z2[2][16 * K2];  // [h1 | h2]
  __shared__ __align__(16) _Float16 zf[16 * KF];     // h2 for FC
  __shared__ float sb[4*144 + 4*96 + 4*32 + 32];     // staged biases (padded)

  const int tid  = threadIdx.x;
  const int lane = tid & 31;
  const int wave = tid >> 5;
  const int row0 = blockIdx.x * 16;
  const int r16  = lane & 15;
  const int g    = lane >> 4;

  // zero state + pads (h(-1)=0; pad cols stay 0 forever)
  for (int i = tid; i < 2 * 16 * K0; i += NTHREADS) ((_Float16*)z0)[i] = (_Float16)0.0f;
  for (int i = tid; i < 2 * 16 * K1; i += NTHREADS) ((_Float16*)z1)[i] = (_Float16)0.0f;
  for (int i = tid; i < 2 * 16 * K2; i += NTHREADS) ((_Float16*)z2)[i] = (_Float16)0.0f;
  for (int i = tid; i < 16 * KF;     i += NTHREADS) zf[i] = (_Float16)0.0f;
  // biases (padded with zeros)
  for (int i = tid; i < 4*144; i += NTHREADS) { int a=i/144, n=i%144; sb[i]       = (n<U0)?P.b[a][n]:0.0f; }
  for (int i = tid; i < 4*96;  i += NTHREADS) { int a=i/96,  n=i%96;  sb[576+i]   = (n<U1)?P.b[4+a][n]:0.0f; }
  for (int i = tid; i < 4*32;  i += NTHREADS) { int a=i/32,  n=i%32;  sb[960+i]   = P.b[8+a][n]; }
  for (int i = tid; i < 32;    i += NTHREADS) { sb[1088+i] = P.b[12][i]; }
  // stage x(0) into z0[0]
  for (int i = tid; i < 16 * IN_F; i += NTHREADS) {
    int r = i >> 7, c = i & (IN_F - 1);
    z0[0][r * K0 + c] = (_Float16)P.x[(((size_t)(row0 + r)) * T_STEPS) * IN_F + c];
  }
  __syncthreads();

  for (int t = 0; t < T_STEPS; ++t) {
    const int p = t & 1;
    const int q = p ^ 1;

    // ---- Phase A: layer0 (inter), 9 N-tiles on 9 waves, K=288 ----
    {
      v8f acc[4] = {};
      const size_t tb = (size_t)wave * (NKC0 * FRAG_HALVES);
      tile_mm4<NKC0, K0>(z0[p], P.w[0]+tb, P.w[1]+tb, P.w[2]+tb, P.w[3]+tb, lane, acc);
      const int n = wave * 16 + r16;
      const float bf1 = sb[n], bf2 = sb[144+n], bta = sb[288+n], btb = sb[432+n];
      const bool on = n < U0;
#pragma unroll
      for (int j = 0; j < 8; ++j) {
        const int m = g * 8 + j;
        float f1 = tanhf(acc[0][j] + bf1);
        float f2 = tanhf(acc[1][j] + bf2);
        float s  = sigmoidf_(acc[2][j] + bta + acc[3][j] + btb);
        float h  = f1 * (1.0f - s) + s * f2;
        if (on) {
          _Float16 hh = (_Float16)h;
          z0[q][m * K0 + IN_F + n] = hh;   // recurrent input for A(t+1)
          z1[p][m * K1 + n]        = hh;   // feed B(t)
        }
      }
    }
    __syncthreads();

    // ---- Phase B: layer1 (command) on waves 0..5, K=224; waves 6..8 stage x(t+1) ----
    if (wave < NT1) {
      v8f acc[4] = {};
      const size_t tb = (size_t)wave * (NKC1 * FRAG_HALVES);
      tile_mm4<NKC1, K1>(z1[p], P.w[4]+tb, P.w[5]+tb, P.w[6]+tb, P.w[7]+tb, lane, acc);
      const int n = wave * 16 + r16;
      const float bf1 = sb[576+n], bf2 = sb[672+n], bta = sb[768+n], btb = sb[864+n];
      const bool on = n < U1;
#pragma unroll
      for (int j = 0; j < 8; ++j) {
        const int m = g * 8 + j;
        float f1 = tanhf(acc[0][j] + bf1);
        float f2 = tanhf(acc[1][j] + bf2);
        float s  = sigmoidf_(acc[2][j] + bta + acc[3][j] + btb);
        float h  = f1 * (1.0f - s) + s * f2;
        if (on) {
          _Float16 hh = (_Float16)h;
          z1[q][m * K1 + U0 + n] = hh;     // recurrent input for B(t+1)
          z2[p][m * K2 + n]      = hh;     // feed C(t)
        }
      }
    } else {
      const int tt = t + 1;
      if (tt < T_STEPS) {
        for (int i = tid - 192; i < 16 * IN_F; i += 96) {
          const int r = i >> 7, c = i & (IN_F - 1);
          z0[q][r * K0 + c] =
            (_Float16)P.x[(((size_t)(row0 + r)) * T_STEPS + tt) * IN_F + c];
        }
      }
    }
    __syncthreads();

    // ---- Phase C: layer2 (motor) on waves 0..1, K=128 ----
    if (wave < NT2) {
      v8f acc[4] = {};
      const size_t tb = (size_t)wave * (NKC2 * FRAG_HALVES);
      tile_mm4<NKC2, K2>(z2[p], P.w[8]+tb, P.w[9]+tb, P.w[10]+tb, P.w[11]+tb, lane, acc);
      const int n = wave * 16 + r16;     // 0..31, always valid
      const float bf1 = sb[960+n], bf2 = sb[992+n], bta = sb[1024+n], btb = sb[1056+n];
#pragma unroll
      for (int j = 0; j < 8; ++j) {
        const int m = g * 8 + j;
        float f1 = tanhf(acc[0][j] + bf1);
        float f2 = tanhf(acc[1][j] + bf2);
        float s  = sigmoidf_(acc[2][j] + bta + acc[3][j] + btb);
        float h  = f1 * (1.0f - s) + s * f2;
        _Float16 hh = (_Float16)h;
        z2[q][m * K2 + U1 + n] = hh;     // recurrent input for C(t+1)
        zf[m * KF + n]         = hh;     // feed FC
      }
    }
    __syncthreads();

    // ---- Phase D: output FC (pool is identity, MOT==OUT), waves 0..1, K=32 ----
    // Overlaps with A(t+1) on other waves: touches only zf (stale-safe) + global out.
    if (wave < NTF) {
      v8f acc = {};
      v16h a = load_a_frag(zf + r16 * KF, 0, g);
      const _Float16* wb = P.w[12] + (size_t)wave * FRAG_HALVES + lane * 16;
      acc = wmma16(a, *(const v16h*)wb, acc);
      const int n = wave * 16 + r16;
      const float bb = sb[1088 + n];
#pragma unroll
      for (int j = 0; j < 8; ++j) {
        const int m = g * 8 + j;
        P.out[(((size_t)(row0 + m)) * T_STEPS + t) * OUTF + n] = acc[j] + bb;
      }
    }
    // no barrier: D reads only zf/global; A(t+1) LDS regions are disjoint.
  }
}

// ---------------------------------------------------------------------------
extern "C" void kernel_launch(void* const* d_in, const int* in_sizes, int n_in,
                              void* d_out, int out_size, void* d_ws, size_t ws_size,
                              hipStream_t stream)
{
  (void)in_sizes; (void)n_in; (void)out_size; (void)ws_size;

  struct MatCfg { int w_idx, b_idx, m_idx, units, cat, ntiles, nkc; };
  // d_in order: x, [l{li}: ff1_w, ff1_b, ff2_w, ff2_b, ta_w, ta_b, tb_w, tb_b, mask] x3, fc_w, fc_b
  static const MatCfg cfg[13] = {
    { 1,  2,  9, U0, 263, 9, NKC0},   // l0 ff1 (masked)
    { 3,  4,  9, U0, 263, 9, NKC0},   // l0 ff2 (masked)
    { 5,  6, -1, U0, 263, 9, NKC0},   // l0 ta
    { 7,  8, -1, U0, 263, 9, NKC0},   // l0 tb
    {10, 11, 18, U1, 224, 6, NKC1},   // l1 ff1 (masked)
    {12, 13, 18, U1, 224, 6, NKC1},   // l1 ff2 (masked)
    {14, 15, -1, U1, 224, 6, NKC1},   // l1 ta
    {16, 17, -1, U1, 224, 6, NKC1},   // l1 tb
    {19, 20, 27, U2, 121, 2, NKC2},   // l2 ff1 (masked)
    {21, 22, 27, U2, 121, 2, NKC2},   // l2 ff2 (masked)
    {23, 24, -1, U2, 121, 2, NKC2},   // l2 ta
    {25, 26, -1, U2, 121, 2, NKC2},   // l2 tb
    {28, 29, -1, OUTF, 32, 2, 1},     // fc
  };

  LNNParams P;
  P.x   = (const float*)d_in[0];
  P.out = (float*)d_out;

  _Float16* wsh = (_Float16*)d_ws;
  size_t off = 0;
  for (int m = 0; m < 13; ++m) {
    const int total = cfg[m].ntiles * cfg[m].nkc * FRAG_HALVES;
    const float* W = (const float*)d_in[cfg[m].w_idx];
    const int* M = (cfg[m].m_idx >= 0) ? (const int*)d_in[cfg[m].m_idx] : nullptr;
    _Float16* dst = wsh + off;
    lnn_prep<<<(total + 255) / 256, 256, 0, stream>>>(
        W, M, dst, cfg[m].units, cfg[m].cat, cfg[m].ntiles, cfg[m].nkc);
    P.w[m] = dst;
    P.b[m] = (const float*)d_in[cfg[m].b_idx];
    off += (size_t)total;
  }

  lnn_main<<<16, NTHREADS, 0, stream>>>(P);
}
